// SeqLSTM_72198400246321
// MI455X (gfx1250) — compile-verified, matrix operand order checked by reading, not run on hardware
//
#include <hip/hip_runtime.h>
#include <cstdint>
#include <cstddef>

// ---------------------------------------------------------------------------
// Types for CDNA5 WMMA
// ---------------------------------------------------------------------------
typedef __attribute__((ext_vector_type(16))) __bf16 v16bf;
typedef __attribute__((ext_vector_type(8)))  float  v8f;
typedef __attribute__((ext_vector_type(4)))  int    v4i;

static constexpr int HID   = 128;      // hidden size
static constexpr int G4    = 512;      // 4*H gate width
static constexpr int LPOS  = 300000;   // total positions
static constexpr int NMB_C = 2000;     // pos-level batch (sequences)
static constexpr int SPOS  = 150;      // pos-level seq length
static constexpr int BMB_C = 21;       // mb-level batch
static constexpr int SMB_C = 100;      // mb-level seq length
static constexpr int TMEGA = 30;       // mega-level seq length

// Async global->LDS (CDNA5): guarded so missing builtins degrade to the
// synchronous staging path. Signature (probe-confirmed via diagnostics):
//   void(..._b128)(int4 addrspace(1)* src, int4 addrspace(3)* dst,
//                  imm int offset, imm int cpol)
#if defined(__has_builtin)
#if __has_builtin(__builtin_amdgcn_global_load_async_to_lds_b128)
#define HAVE_ASYNC_LDS 1
#endif
#endif
#ifndef HAVE_ASYNC_LDS
#define HAVE_ASYNC_LDS 0
#endif

#if HAVE_ASYNC_LDS
typedef __attribute__((address_space(1))) v4i  g_v4i;   // global int4
typedef __attribute__((address_space(3))) v4i  l_v4i;   // LDS int4
typedef __attribute__((address_space(3))) char as3_char;
#endif

// ---------------------------------------------------------------------------
// Helpers
// ---------------------------------------------------------------------------
__device__ __forceinline__ unsigned short f32_to_bf16(float f) {
  unsigned u = __float_as_uint(f);
  u += 0x7FFFu + ((u >> 16) & 1u);          // round-to-nearest-even
  return (unsigned short)(u >> 16);
}

union Frag16 { uint4 q[2]; v16bf v; };

// A-matrix fragment (16x32 bf16) from a row-major [16][128] bf16 tile.
__device__ __forceinline__ v16bf load_a_frag(const unsigned short* sh, int row, int khalf, int kb) {
  const unsigned short* p = sh + row * HID + kb * 32 + khalf * 8;
  Frag16 f;
  f.q[0] = *(const uint4*)(p);        // K offsets 0..7
  f.q[1] = *(const uint4*)(p + 16);   // K offsets 16..23
  return f.v;
}

// B-matrix fragment: pre-repacked so each lane's 16 bf16 values are contiguous.
__device__ __forceinline__ v16bf load_b_frag(const unsigned short* base, int nt, int kb, int lane) {
  const unsigned short* p = base + (size_t)(((nt << 2) + kb) * 32 + lane) * 16;
  Frag16 f;
  f.q[0] = *(const uint4*)(p);
  f.q[1] = *(const uint4*)(p + 8);
  return f.v;
}

__device__ __forceinline__ v8f wmma_bf16(v16bf a, v16bf b, v8f c) {
  return __builtin_amdgcn_wmma_f32_16x16x32_bf16(false, a, false, b, (short)0, c, false, false);
}

__device__ __forceinline__ float sigmoidf_(float x) { return 1.0f / (1.0f + __expf(-x)); }

// ---------------------------------------------------------------------------
// Weight repack: W (512x128 f32, row-major, raw = x @ W^T) -> WMMA B-fragment
// layout in bf16: frag[(nt*4+kb)*32 + lane][e], B[k][n] = W[n][k].
// ---------------------------------------------------------------------------
__global__ void repack_w_kernel(const float* __restrict__ W, unsigned short* __restrict__ dst) {
  int i = blockIdx.x * 256 + threadIdx.x;        // 0 .. 65535
  if (i >= G4 * HID) return;
  int e     = i & 15;
  int lane  = (i >> 4) & 31;
  int kb    = (i >> 9) & 3;
  int nt    = i >> 11;
  int n     = nt * 16 + (lane & 15);
  int khalf = lane >> 4;
  int k     = kb * 32 + khalf * 8 + ((e < 8) ? e : (e + 8));
  dst[i] = f32_to_bf16(W[n * HID + k]);
}

// ---------------------------------------------------------------------------
// Embedding gather -> bf16 activations
// ---------------------------------------------------------------------------
__global__ void gather_x_kernel(const int* __restrict__ idx, const float* __restrict__ embed,
                                unsigned short* __restrict__ xbf, int total) {
  int i = blockIdx.x * 256 + threadIdx.x;
  if (i >= total) return;
  int row = i >> 7, h = i & 127;
  xbf[i] = f32_to_bf16(embed[(size_t)idx[row] * HID + h]);
}

// Average final hidden states of both directions -> next-level bf16 input (zero pad)
__global__ void avg_ht_kernel(const float* __restrict__ a, const float* __restrict__ b,
                              unsigned short* __restrict__ x, int nvalid, int ntotal) {
  int i = blockIdx.x * 256 + threadIdx.x;
  if (i >= ntotal * HID) return;
  int r = i >> 7;
  float v = (r < nvalid) ? 0.5f * (a[i] + b[i]) : 0.0f;
  x[i] = f32_to_bf16(v);
}

// ---------------------------------------------------------------------------
// LayerNorm-LSTM, one direction. Workgroup = 8 waves = 256 threads, handles a
// 16-sequence batch tile across T timesteps. gates = LN(Wih x) + LN(Whh h),
// c = LN(sig(f)*c + sig(i)*tanh(g)), h = sig(o)*tanh(c).
// Whh WMMA fragments are LDS-resident (read 150x); Wih fragments stream from
// L2; x_t tiles are staged memory->LDS with CDNA5 async loads (ASYNCcnt).
// ---------------------------------------------------------------------------
__launch_bounds__(256)
__global__ void lnlstm_dir_kernel(const unsigned short* __restrict__ xbf,
                                  const unsigned short* __restrict__ wih_frag,
                                  const unsigned short* __restrict__ whh_frag,
                                  const float* __restrict__ lig, const float* __restrict__ lib_,
                                  const float* __restrict__ lhg, const float* __restrict__ lhb,
                                  const float* __restrict__ lcg, const float* __restrict__ lcb,
                                  const float* __restrict__ h0, const float* __restrict__ c0,
                                  float* __restrict__ out_h, float* __restrict__ hT,
                                  int B, int T, int reverse) {
  __shared__ alignas(16) unsigned short sh_whh[G4 * HID];    // 128 KB (Whh frags)
  __shared__ alignas(16) unsigned short sh_x[16 * HID];      // 4 KB  (x_t bf16)
  __shared__ alignas(16) unsigned short sh_h[16 * HID];      // 4 KB  (h bf16)
  __shared__ float sh_gI[16][G4 + 8];                        // 33 KB (raw Wih·x)
  __shared__ float sh_gH[16][G4 + 8];                        // 33 KB (raw Whh·h)
  __shared__ float sh_c[16][HID];                            // 8 KB
  __shared__ float sh_cr[16][HID + 4];                       // pre-LN cell
  __shared__ float sh_hf[16][HID];                           // h in f32
  __shared__ float sh_st[16][8];                             // LN stats

  const int tid   = threadIdx.x;
  const int lane  = tid & 31;
  const int wave  = tid >> 5;
  const int row0  = blockIdx.x * 16;
  const int khalf = lane >> 4;
  const int arow  = lane & 15;

  // Whh fragments -> LDS, once (8192 x uint4 = 128 KB)
  {
    const uint4* src = (const uint4*)whh_frag;
    uint4* dst = (uint4*)sh_whh;
    for (int i = tid; i < (G4 * HID) / 8; i += 256) dst[i] = src[i];
  }
  // init h, c; pre-zero pad rows of sh_x (async path only writes valid rows)
  for (int i = tid; i < 16 * HID; i += 256) {
    int r = i >> 7, h = i & 127;
    int seq = row0 + r;
    float hv = (seq < B) ? h0[(size_t)seq * HID + h] : 0.0f;
    float cv = (seq < B) ? c0[(size_t)seq * HID + h] : 0.0f;
    sh_h[i]     = f32_to_bf16(hv);
    sh_hf[r][h] = hv;
    sh_c[r][h]  = cv;
    if (seq >= B) sh_x[i] = 0;
  }
  __syncthreads();

  for (int s = 0; s < T; ++s) {
    const int t = reverse ? (T - 1 - s) : s;

    // --- stage x_t into LDS (16 rows x 128 bf16) ---
#if HAVE_ASYNC_LDS
    {
      // one b128 per thread: row = tid>>4, 16B chunk = tid&15
      int r = tid >> 4, chunk = tid & 15;
      int seq = row0 + r;
      if (seq < B) {
        const unsigned short* gp = xbf + ((size_t)seq * T + t) * HID + chunk * 8;
        as3_char* lp = (as3_char*)sh_x + tid * 16;
        __builtin_amdgcn_global_load_async_to_lds_b128((g_v4i*)gp, (l_v4i*)lp, 0, 0);
      }
      asm volatile("s_wait_asynccnt 0x0" ::: "memory");
    }
#else
    for (int i = tid; i < 16 * HID / 2; i += 256) {   // dword granularity
      int r = i >> 6, cpair = i & 63;
      int seq = row0 + r;
      unsigned v = 0u;
      if (seq < B) v = ((const unsigned*)(xbf + ((size_t)seq * T + t) * HID))[cpair];
      ((unsigned*)sh_x)[i] = v;
    }
#endif
    // prefetch next timestep's rows into cache (global_prefetch_b8)
    if (s + 1 < T && tid < 16) {
      int tn  = reverse ? (T - 2 - s) : (s + 1);
      int seq = row0 + tid;
      if (seq < B) __builtin_prefetch(xbf + ((size_t)seq * T + tn) * HID, 0, 1);
    }
    __syncthreads();

    // --- two GEMMs: raw_ig = x @ Wih^T, raw_hg = h @ Whh^T (M=16,N=512,K=128)
    v8f accI[4], accH[4];
#pragma unroll
    for (int j = 0; j < 4; ++j) {
      v8f z = {0.f, 0.f, 0.f, 0.f, 0.f, 0.f, 0.f, 0.f};
      accI[j] = z; accH[j] = z;
    }
#pragma unroll
    for (int kb = 0; kb < 4; ++kb) {
      v16bf ax = load_a_frag(sh_x, arow, khalf, kb);
      v16bf ah = load_a_frag(sh_h, arow, khalf, kb);
#pragma unroll
      for (int j = 0; j < 4; ++j) {
        int nt = wave * 4 + j;
        v16bf bi = load_b_frag(wih_frag, nt, kb, lane);   // L2-resident stream
        v16bf bh = load_b_frag(sh_whh,  nt, kb, lane);    // LDS-resident
        accI[j] = wmma_bf16(ax, bi, accI[j]);
        accH[j] = wmma_bf16(ah, bh, accH[j]);
      }
    }
    // scatter C fragments: lane<16 -> rows 0..7, lane>=16 -> rows 8..15
    {
      int rbase = (lane < 16) ? 0 : 8;
#pragma unroll
      for (int j = 0; j < 4; ++j) {
        int col = (wave * 4 + j) * 16 + arow;
#pragma unroll
        for (int r = 0; r < 8; ++r) {
          sh_gI[rbase + r][col] = accI[j][r];
          sh_gH[rbase + r][col] = accH[j][r];
        }
      }
    }
    __syncthreads();

    // --- LN stats over 512 gate cols, per row (wave handles 2 rows) ---
#pragma unroll
    for (int rr = 0; rr < 2; ++rr) {
      int r = wave * 2 + rr;
      float sI = 0.f, sI2 = 0.f, sH = 0.f, sH2 = 0.f;
#pragma unroll
      for (int k = 0; k < 16; ++k) {
        float a = sh_gI[r][lane + 32 * k];
        float b = sh_gH[r][lane + 32 * k];
        sI += a; sI2 += a * a; sH += b; sH2 += b * b;
      }
#pragma unroll
      for (int off = 16; off > 0; off >>= 1) {
        sI  += __shfl_xor(sI,  off, 32);
        sI2 += __shfl_xor(sI2, off, 32);
        sH  += __shfl_xor(sH,  off, 32);
        sH2 += __shfl_xor(sH2, off, 32);
      }
      if (lane == 0) {
        float muI = sI * (1.f / 512.f), muH = sH * (1.f / 512.f);
        float vI = sI2 * (1.f / 512.f) - muI * muI;
        float vH = sH2 * (1.f / 512.f) - muH * muH;
        sh_st[r][0] = muI; sh_st[r][1] = rsqrtf(vI + 1e-5f);
        sh_st[r][2] = muH; sh_st[r][3] = rsqrtf(vH + 1e-5f);
      }
    }
    __syncthreads();

    // --- gates + cell pre-activation ---
    float o_save[8];
#pragma unroll
    for (int k = 0; k < 8; ++k) {
      int idx = tid + 256 * k;
      int r = idx >> 7, h = idx & 127;
      float muI = sh_st[r][0], rsI = sh_st[r][1];
      float muH = sh_st[r][2], rsH = sh_st[r][3];
      auto gv = [&](int c) {
        return (sh_gI[r][c] - muI) * rsI * lig[c] + lib_[c] +
               (sh_gH[r][c] - muH) * rsH * lhg[c] + lhb[c];
      };
      float gi = gv(h);
      float gf = gv(128 + h);
      float gg = gv(256 + h);
      float go = gv(384 + h);
      sh_cr[r][h] = sigmoidf_(gf) * sh_c[r][h] + sigmoidf_(gi) * tanhf(gg);
      o_save[k] = go;
    }
    __syncthreads();

    // --- cell LN stats over 128, per row ---
#pragma unroll
    for (int rr = 0; rr < 2; ++rr) {
      int r = wave * 2 + rr;
      float s = 0.f, s2 = 0.f;
#pragma unroll
      for (int k = 0; k < 4; ++k) {
        float a = sh_cr[r][lane + 32 * k];
        s += a; s2 += a * a;
      }
#pragma unroll
      for (int off = 16; off > 0; off >>= 1) {
        s  += __shfl_xor(s,  off, 32);
        s2 += __shfl_xor(s2, off, 32);
      }
      if (lane == 0) {
        float mu = s * (1.f / 128.f);
        float v  = s2 * (1.f / 128.f) - mu * mu;
        sh_st[r][4] = mu; sh_st[r][5] = rsqrtf(v + 1e-5f);
      }
    }
    __syncthreads();

    // --- finalize c, h; emit h_t ---
#pragma unroll
    for (int k = 0; k < 8; ++k) {
      int idx = tid + 256 * k;
      int r = idx >> 7, h = idx & 127;
      float mu = sh_st[r][4], rs = sh_st[r][5];
      float c = (sh_cr[r][h] - mu) * rs * lcg[h] + lcb[h];
      sh_c[r][h] = c;
      float hv = sigmoidf_(o_save[k]) * tanhf(c);
      sh_hf[r][h]       = hv;
      sh_h[r * HID + h] = f32_to_bf16(hv);
      int seq = row0 + r;
      if (seq < B) out_h[((size_t)seq * T + t) * HID + h] = hv;
    }
    __syncthreads();
  }

  // final hidden state
  for (int i = tid; i < 16 * HID; i += 256) {
    int r = i >> 7, h = i & 127;
    int seq = row0 + r;
    if (seq < B) hT[(size_t)seq * HID + h] = sh_hf[r][h];
  }
}

// ---------------------------------------------------------------------------
// Final assembly: out row 0 = zero_embed, rows 1..nrows = concat(pos, mb, mega)
// pair_avg implements reshape(-1,H,2).mean(2): adjacent-channel pair average.
// ---------------------------------------------------------------------------
__device__ __forceinline__ float pair_avg(const float* of, const float* ob, int r, int h) {
  if (h < 64) {
    size_t base = (size_t)r * HID + 2 * h;
    return 0.5f * (of[base] + of[base + 1]);
  }
  size_t base = (size_t)r * HID + 2 * (h - 64);
  return 0.5f * (ob[base] + ob[base + 1]);
}

__global__ void combine_kernel(const float* __restrict__ pf, const float* __restrict__ pb,
                               const float* __restrict__ mf, const float* __restrict__ mb_,
                               const float* __restrict__ gf, const float* __restrict__ gb,
                               float* __restrict__ out, int nrows) {
  long long i = (long long)blockIdx.x * 256 + threadIdx.x;
  long long total = (long long)(nrows + 1) * 384;
  if (i >= total) return;
  int col = (int)(i % 384);
  long long row = i / 384;
  int h = col & 127, sec = col >> 7;
  int pr, mr, gr;
  if (row == 0) { pr = LPOS - 1; mr = BMB_C * SMB_C - 1; gr = TMEGA - 1; }
  else {
    long long q = row - 1;
    pr = (int)q;
    mr = (int)(q / SPOS);
    gr = (int)(q / ((long long)SPOS * SMB_C));
  }
  float v;
  if      (sec == 0) v = pair_avg(pf, pb, pr, h);
  else if (sec == 1) v = pair_avg(mf, mb_, mr, h);
  else               v = pair_avg(gf, gb, gr, h);
  out[i] = v;
}

// ---------------------------------------------------------------------------
// Host launcher
// ---------------------------------------------------------------------------
extern "C" void kernel_launch(void* const* d_in, const int* in_sizes, int n_in,
                              void* d_out, int out_size, void* d_ws, size_t ws_size,
                              hipStream_t stream) {
  (void)in_sizes; (void)n_in; (void)ws_size;

  const int*   idx   = (const int*)d_in[0];
  const float* embed = (const float*)d_in[2];

  const float* WihL[3] = {(const float*)d_in[3],  (const float*)d_in[11], (const float*)d_in[19]};
  const float* WhhL[3] = {(const float*)d_in[4],  (const float*)d_in[12], (const float*)d_in[20]};
  const float* ligL[3] = {(const float*)d_in[5],  (const float*)d_in[13], (const float*)d_in[21]};
  const float* libL[3] = {(const float*)d_in[6],  (const float*)d_in[14], (const float*)d_in[22]};
  const float* lhgL[3] = {(const float*)d_in[7],  (const float*)d_in[15], (const float*)d_in[23]};
  const float* lhbL[3] = {(const float*)d_in[8],  (const float*)d_in[16], (const float*)d_in[24]};
  const float* lcgL[3] = {(const float*)d_in[9],  (const float*)d_in[17], (const float*)d_in[25]};
  const float* lcbL[3] = {(const float*)d_in[10], (const float*)d_in[18], (const float*)d_in[26]};
  const float* h0L[3]  = {(const float*)d_in[27], (const float*)d_in[29], (const float*)d_in[31]};
  const float* c0L[3]  = {(const float*)d_in[28], (const float*)d_in[30], (const float*)d_in[32]};
  const int    Blv[3]  = {NMB_C, BMB_C, 1};
  const int    Tlv[3]  = {SPOS, SMB_C, TMEGA};

  const int nrows = out_size / 384 - 1;

  // workspace carve-up
  char* p = (char*)d_ws;
  auto take = [&](size_t n) { char* r = p; p += (n + 255) & ~(size_t)255; return r; };
  unsigned short* xbf = (unsigned short*)take((size_t)LPOS * HID * 2);
  unsigned short* wfr = (unsigned short*)take((size_t)12 * G4 * HID * 2);
  float* pos_of  = (float*)take((size_t)LPOS * HID * 4);
  float* pos_ob  = (float*)take((size_t)LPOS * HID * 4);
  float* pos_hTf = (float*)take((size_t)NMB_C * HID * 4);
  float* pos_hTb = (float*)take((size_t)NMB_C * HID * 4);
  unsigned short* mbx = (unsigned short*)take((size_t)BMB_C * SMB_C * HID * 2);
  float* mb_of   = (float*)take((size_t)BMB_C * SMB_C * HID * 4);
  float* mb_ob   = (float*)take((size_t)BMB_C * SMB_C * HID * 4);
  float* mb_hTf  = (float*)take((size_t)BMB_C * HID * 4);
  float* mb_hTb  = (float*)take((size_t)BMB_C * HID * 4);
  unsigned short* mgx = (unsigned short*)take((size_t)TMEGA * HID * 2);
  float* mg_of   = (float*)take((size_t)TMEGA * HID * 4);
  float* mg_ob   = (float*)take((size_t)TMEGA * HID * 4);
  float* mg_hTf  = (float*)take((size_t)HID * 4);
  float* mg_hTb  = (float*)take((size_t)HID * 4);

  // 1) repack all weights into WMMA B-fragment bf16 layout (tiny, L2-resident)
  auto WF = [&](int lv, int which, int dir) {
    return wfr + (size_t)(lv * 4 + dir * 2 + which) * (G4 * HID);
  };
  for (int lv = 0; lv < 3; ++lv)
    for (int dir = 0; dir < 2; ++dir) {
      repack_w_kernel<<<256, 256, 0, stream>>>(WihL[lv] + (size_t)dir * G4 * HID, WF(lv, 0, dir));
      repack_w_kernel<<<256, 256, 0, stream>>>(WhhL[lv] + (size_t)dir * G4 * HID, WF(lv, 1, dir));
    }

  // 2) embedding gather -> bf16
  {
    int total = LPOS * HID;
    gather_x_kernel<<<(total + 255) / 256, 256, 0, stream>>>(idx, embed, xbf, total);
  }

  // 3) run the three bidirectional levels
  auto run_level = [&](int lv, const unsigned short* x,
                       float* of, float* ob, float* hTf, float* hTb) {
    int B = Blv[lv], T = Tlv[lv];
    int tiles = (B + 15) / 16;
    for (int dir = 0; dir < 2; ++dir) {
      lnlstm_dir_kernel<<<tiles, 256, 0, stream>>>(
          x, WF(lv, 0, dir), WF(lv, 1, dir),
          ligL[lv] + (size_t)dir * G4, libL[lv] + (size_t)dir * G4,
          lhgL[lv] + (size_t)dir * G4, lhbL[lv] + (size_t)dir * G4,
          lcgL[lv] + (size_t)dir * HID, lcbL[lv] + (size_t)dir * HID,
          h0L[lv] + (size_t)dir * B * HID, c0L[lv] + (size_t)dir * B * HID,
          dir ? ob : of, dir ? hTb : hTf, B, T, dir);
    }
  };

  run_level(0, xbf, pos_of, pos_ob, pos_hTf, pos_hTb);

  {
    int total = BMB_C * SMB_C * HID;
    avg_ht_kernel<<<(total + 255) / 256, 256, 0, stream>>>(pos_hTf, pos_hTb, mbx,
                                                           NMB_C, BMB_C * SMB_C);
  }
  run_level(1, mbx, mb_of, mb_ob, mb_hTf, mb_hTb);

  {
    int total = TMEGA * HID;
    avg_ht_kernel<<<(total + 255) / 256, 256, 0, stream>>>(mb_hTf, mb_hTb, mgx,
                                                           BMB_C, TMEGA);
  }
  run_level(2, mgx, mg_of, mg_ob, mg_hTf, mg_hTb);

  // 4) final assembly
  {
    long long total = (long long)(nrows + 1) * 384;
    int blocks = (int)((total + 255) / 256);
    combine_kernel<<<blocks, 256, 0, stream>>>(pos_of, pos_ob, mb_of, mb_ob,
                                               mg_of, mg_ob, (float*)d_out, nrows);
  }
}
